// LightweightCrossAttention_18150531792901
// MI455X (gfx1250) — matrix operand order repaired and checked by reading, not compiled, not run on hardware
//
#include <hip/hip_runtime.h>

typedef _Float16 v16h  __attribute__((ext_vector_type(16)));
typedef _Float16 v8h   __attribute__((ext_vector_type(8)));
typedef float    v8f   __attribute__((ext_vector_type(8)));
typedef float    v8f32 __attribute__((ext_vector_type(8)));

#define WMMA_F32_F16(a, b, c) \
    __builtin_amdgcn_wmma_f32_16x16x32_f16(false, (a), false, (b), (short)0, (c), false, false)

// Problem constants (fixed by the reference): B=8, Tq=Tc=2048, D=256
constexpr int   kB     = 8;
constexpr int   kT     = 2048;       // Tq == Tc
constexpr int   kD     = 256;
constexpr int   kM     = kB * kT;    // 16384 rows for both Q and KV
constexpr float kScale = 0.0625f;    // D^-0.5

// ---- CDNA5 WMMA 16x16x32 f16 fragment layouts (wave32) ----
// A (16x32, MxK): lane&15 = M row; lane half h selects K sub-block:
//   per-lane data = halves [k0 + h*8 .. +7] and [k0 + 16 + h*8 .. +7] (2x b128)
// B (32x16, KxN): lane&15 = N col; per-lane data = halves [k0 + h*8 .. +7] and [k0 + 16 + h*8 .. +7] (2x b128)
// C/D (16x16 f32): element v -> row m = v + 8*h, col n = lane&15

__device__ __forceinline__ v16h load_a_f16(const _Float16* rowptr, int k0, int h) {
    const v8h lo = *(const v8h*)(rowptr + k0 + h * 8);
    const v8h hi = *(const v8h*)(rowptr + k0 + 16 + h * 8);
    return __builtin_shufflevector(lo, hi, 0, 1, 2, 3, 4, 5, 6, 7,
                                           8, 9, 10, 11, 12, 13, 14, 15);
}
__device__ __forceinline__ v16h load_b_f16(const _Float16* rowptr, int k0, int h) {
    const v8h lo = *(const v8h*)(rowptr + k0 + h * 8);
    const v8h hi = *(const v8h*)(rowptr + k0 + 16 + h * 8);
    return __builtin_shufflevector(lo, hi, 0, 1, 2, 3, 4, 5, 6, 7,
                                           8, 9, 10, 11, 12, 13, 14, 15);
}

// ============================================================================
// Kernel 0: one-pass fp32 -> f16 conversion (b128 in, packed cvt, b128 out).
// n8 = number of 8-element groups.
// ============================================================================
__global__ __launch_bounds__(256, 1)
void cvt_f32_to_f16(const float* __restrict__ src, _Float16* __restrict__ dst, int n8)
{
    const int i = blockIdx.x * 256 + threadIdx.x;
    if (i >= n8) return;
    const v8f32 x = *(const v8f32*)(src + (long)i * 8);
    v8h y;
#pragma unroll
    for (int j = 0; j < 8; ++j) y[j] = (_Float16)x[j];
    *(v8h*)(dst + (long)i * 8) = y;
}

// ============================================================================
// Kernel 1a: Q projection. Qf[m,n] = sum_k Ah[m,k]*Wh[n,k] + bias[n]  (f16)
//   All-f16 operands: pure b128 loads + wmma inner loop.
// ============================================================================
__global__ __launch_bounds__(256, 1)
void proj_q_kernel(const _Float16* __restrict__ A, const _Float16* __restrict__ W,
                   const float* __restrict__ bias, _Float16* __restrict__ out)
{
    const int lane = threadIdx.x & 31;
    const int wv   = threadIdx.x >> 5;
    const int h    = lane >> 4;
    const int ln   = lane & 15;

    const int tile = blockIdx.x * 8 + wv;          // 16384 tiles
    const int mt   = tile >> 4;                    // ntiles = 16
    const int nt   = tile & 15;

    const _Float16* arow = A + (long)(mt * 16 + ln) * kD;
    const _Float16* wrow = W + (long)(nt * 16 + ln) * kD;

    v8f c = {};
#pragma unroll
    for (int k0 = 0; k0 < kD; k0 += 32)
        c = WMMA_F32_F16(load_a_f16(arow, k0, h), load_b_f16(wrow, k0, h), c);

    const float bn = bias[nt * 16 + ln];
#pragma unroll
    for (int v = 0; v < 8; ++v) {
        const int m = mt * 16 + v + 8 * h;
        out[(long)m * kD + nt * 16 + ln] = (_Float16)(c[v] + bn);
    }
}

// ============================================================================
// Kernel 1b: KV projection. N=512: cols 0..255 -> K row-major f16,
//            cols 256..511 -> V transposed (d-major): Vt[(b*256+d)*2048 + t]
// ============================================================================
__global__ __launch_bounds__(256, 1)
void proj_kv_kernel(const _Float16* __restrict__ A, const _Float16* __restrict__ W,
                    const float* __restrict__ bias,
                    _Float16* __restrict__ outK, _Float16* __restrict__ outVt)
{
    const int lane = threadIdx.x & 31;
    const int wv   = threadIdx.x >> 5;
    const int h    = lane >> 4;
    const int ln   = lane & 15;

    const int tile = blockIdx.x * 8 + wv;          // 32768 tiles
    const int mt   = tile >> 5;                    // ntiles = 32
    const int nt   = tile & 31;

    const _Float16* arow = A + (long)(mt * 16 + ln) * kD;
    const _Float16* wrow = W + (long)(nt * 16 + ln) * kD;

    v8f c = {};
#pragma unroll
    for (int k0 = 0; k0 < kD; k0 += 32)
        c = WMMA_F32_F16(load_a_f16(arow, k0, h), load_b_f16(wrow, k0, h), c);

    const int   ncol = nt * 16 + ln;
    const float bn   = bias[ncol];

    if (ncol < kD) {
        // K part, row-major
#pragma unroll
        for (int v = 0; v < 8; ++v) {
            const int m = mt * 16 + v + 8 * h;
            outK[(long)m * kD + ncol] = (_Float16)(c[v] + bn);
        }
    } else {
        // V part, transposed: m = b*2048 + t  ->  Vt[(b*256 + d)*2048 + t]
        const int d = ncol - kD;
#pragma unroll
        for (int v = 0; v < 8; ++v) {
            const int m  = mt * 16 + v + 8 * h;
            const int bb = m >> 11;                 // / 2048
            const int t  = m & 2047;                // % 2048
            outVt[((long)bb * kD + d) * kT + t] = (_Float16)(c[v] + bn);
        }
    }
}

// ============================================================================
// Kernel 2: flash attention. One wave per 16-row Q tile, 32-key chunks,
// online softmax, O accumulated in f32 (16 x 256 per wave).
// ============================================================================
__global__ __launch_bounds__(256, 1)
void flash_attn(const _Float16* __restrict__ Qf, const _Float16* __restrict__ Kf,
                const _Float16* __restrict__ Vt, _Float16* __restrict__ Of)
{
    __shared__ __align__(16) _Float16 pbuf[8][16][32];  // per-wave P transpose tile

    const int lane = threadIdx.x & 31;
    const int wv   = threadIdx.x >> 5;
    const int h    = lane >> 4;
    const int ln   = lane & 15;

    const int  tile  = blockIdx.x * 8 + wv;        // 1024 wave-tiles
    const long qrow0 = (long)tile * 16;
    const int  bb    = (int)(qrow0 >> 11);         // / Tq (=2048)

    const _Float16* Kb = Kf + (long)bb * kT * kD;
    const _Float16* Vb = Vt + (long)bb * kD * kT;

    // Resident Q fragments: 8 A-frags covering k = 0..255
    const _Float16* qrow = Qf + (qrow0 + ln) * kD;
    v16h qf[8];
#pragma unroll
    for (int j = 0; j < 8; ++j) qf[j] = load_a_f16(qrow, j * 32, h);

    v8f acc[16];
    const v8f vzero = {};
#pragma unroll
    for (int nt = 0; nt < 16; ++nt) acc[nt] = vzero;

    float mrun[8], lrun[8];
#pragma unroll
    for (int v = 0; v < 8; ++v) { mrun[v] = -1e30f; lrun[v] = 0.0f; }

    for (int ch = 0; ch < kT / 32; ++ch) {
        const int key0 = ch * 32;

        // ---- S = Q K^T for 32 keys (two 16x16 tiles) ----
        const _Float16* krow0 = Kb + (long)(key0 + ln) * kD;
        const _Float16* krow1 = Kb + (long)(key0 + 16 + ln) * kD;
        v8f s0 = vzero, s1 = vzero;
#pragma unroll
        for (int j = 0; j < 8; ++j) {
            const int k0 = j * 32;
            s0 = WMMA_F32_F16(qf[j], load_b_f16(krow0, k0, h), s0);
            s1 = WMMA_F32_F16(qf[j], load_b_f16(krow1, k0, h), s1);
        }

        // ---- online softmax (row m = v + 8*h across the 16 lanes of a half) ----
        float p0[8], p1[8], corr[8];
#pragma unroll
        for (int v = 0; v < 8; ++v) {
            const float x0 = s0[v] * kScale;
            const float x1 = s1[v] * kScale;
            float mx = fmaxf(x0, x1);
#pragma unroll
            for (int msk = 1; msk <= 8; msk <<= 1) mx = fmaxf(mx, __shfl_xor(mx, msk, 32));
            const float mnew = fmaxf(mrun[v], mx);
            corr[v] = __expf(mrun[v] - mnew);
            const float e0 = __expf(x0 - mnew);
            const float e1 = __expf(x1 - mnew);
            float rs = e0 + e1;
#pragma unroll
            for (int msk = 1; msk <= 8; msk <<= 1) rs += __shfl_xor(rs, msk, 32);
            lrun[v] = lrun[v] * corr[v] + rs;
            mrun[v] = mnew;
            p0[v] = e0;
            p1[v] = e1;
        }

        // rescale accumulators
#pragma unroll
        for (int nt = 0; nt < 16; ++nt)
#pragma unroll
            for (int v = 0; v < 8; ++v) acc[nt][v] *= corr[v];

        // ---- transpose P (C-layout -> A-layout) via per-wave LDS tile ----
#pragma unroll
        for (int v = 0; v < 8; ++v) {
            pbuf[wv][v + 8 * h][ln]      = (_Float16)p0[v];
            pbuf[wv][v + 8 * h][16 + ln] = (_Float16)p1[v];
        }
        const v16h pf = load_a_f16(&pbuf[wv][ln][0], 0, h);

        // ---- O += P (16x32) x V (32x256) : 16 n-tiles ----
#pragma unroll
        for (int nt = 0; nt < 16; ++nt) {
            const _Float16* vrow = Vb + (long)(nt * 16 + ln) * kT + key0;
            acc[nt] = WMMA_F32_F16(pf, load_b_f16(vrow, 0, h), acc[nt]);
        }
    }

    // ---- epilogue: normalize by l and store O as f16 ----
#pragma unroll
    for (int v = 0; v < 8; ++v) {
        const float inv = 1.0f / lrun[v];
        const long  m   = qrow0 + v + 8 * h;
#pragma unroll
        for (int nt = 0; nt < 16; ++nt)
            Of[m * kD + nt * 16 + ln] = (_Float16)(acc[nt][v] * inv);
    }
}

// ============================================================================
// Kernel 3: output projection, f16 A x f16 W -> fp32 out + bias.
// ============================================================================
__global__ __launch_bounds__(256, 1)
void out_gemm_f32(const _Float16* __restrict__ A, const _Float16* __restrict__ W,
                  const float* __restrict__ bias, float* __restrict__ out)
{
    const int lane = threadIdx.x & 31;
    const int wv   = threadIdx.x >> 5;
    const int h    = lane >> 4;
    const int ln   = lane & 15;

    const int tile = blockIdx.x * 8 + wv;          // 16384 tiles
    const int mt   = tile >> 4;                    // ntiles = 16
    const int nt   = tile & 15;

    const _Float16* arow = A + (long)(mt * 16 + ln) * kD;
    const _Float16* wrow = W + (long)(nt * 16 + ln) * kD;

    v8f c = {};
#pragma unroll
    for (int k0 = 0; k0 < kD; k0 += 32)
        c = WMMA_F32_F16(load_a_f16(arow, k0, h), load_b_f16(wrow, k0, h), c);

    const float bn = bias[nt * 16 + ln];
#pragma unroll
    for (int v = 0; v < 8; ++v) {
        const int m = mt * 16 + v + 8 * h;
        out[(long)m * kD + nt * 16 + ln] = c[v] + bn;
    }
}

// ============================================================================
// Host launcher
// ============================================================================
extern "C" void kernel_launch(void* const* d_in, const int* in_sizes, int n_in,
                              void* d_out, int out_size, void* d_ws, size_t ws_size,
                              hipStream_t stream)
{
    (void)in_sizes; (void)n_in; (void)out_size; (void)ws_size;

    const float* query   = (const float*)d_in[0];
    const float* context = (const float*)d_in[1];
    const float* Wq      = (const float*)d_in[2];
    const float* bq      = (const float*)d_in[3];
    const float* Wkv     = (const float*)d_in[4];
    const float* bkv     = (const float*)d_in[5];
    const float* Wo      = (const float*)d_in[6];
    const float* bo      = (const float*)d_in[7];
    float*       out     = (float*)d_out;

    // Workspace (all f16):
    //   Qf | Kf | Vt | Of            4 x 8 MB
    //   Ah (query) | Ch (context)    2 x 8 MB
    //   Wq16 | Wkv16 | Wo16          128K + 256K + 128K
    _Float16* Qf    = (_Float16*)d_ws;
    _Float16* Kf    = Qf  + (long)kM * kD;
    _Float16* Vt    = Kf  + (long)kM * kD;
    _Float16* Of    = Vt  + (long)kM * kD;
    _Float16* Ah    = Of  + (long)kM * kD;
    _Float16* Ch    = Ah  + (long)kM * kD;
    _Float16* Wq16  = Ch  + (long)kM * kD;
    _Float16* Wkv16 = Wq16  + (long)kD * kD;
    _Float16* Wo16  = Wkv16 + (long)2 * kD * kD;

    // 0) One-pass fp32 -> f16 conversions
    cvt_f32_to_f16<<<dim3(2048), dim3(256), 0, stream>>>(query,   Ah,    kM * kD / 8);
    cvt_f32_to_f16<<<dim3(2048), dim3(256), 0, stream>>>(context, Ch,    kM * kD / 8);
    cvt_f32_to_f16<<<dim3(32),   dim3(256), 0, stream>>>(Wq,      Wq16,  kD * kD / 8);
    cvt_f32_to_f16<<<dim3(64),   dim3(256), 0, stream>>>(Wkv,     Wkv16, 2 * kD * kD / 8);
    cvt_f32_to_f16<<<dim3(32),   dim3(256), 0, stream>>>(Wo,      Wo16,  kD * kD / 8);

    // 1a) Q projection: 16384 tiles / 8 waves = 2048 blocks
    proj_q_kernel<<<dim3(2048), dim3(256), 0, stream>>>(Ah, Wq16, bq, Qf);

    // 1b) KV projection: 32768 tiles / 8 = 4096 blocks
    proj_kv_kernel<<<dim3(4096), dim3(256), 0, stream>>>(Ch, Wkv16, bkv, Kf, Vt);

    // 2) Flash attention: 1024 wave-tiles / 8 = 128 blocks
    flash_attn<<<dim3(128), dim3(256), 0, stream>>>(Qf, Kf, Vt, Of);

    // 3) Output projection: 2048 blocks
    out_gemm_f32<<<dim3(2048), dim3(256), 0, stream>>>(Of, Wo16, bo, out);
}
